// ScannedRNN_7773890806122
// MI455X (gfx1250) — compile-verified, matrix operand order checked by reading, not run on hardware
//
#include <hip/hip_runtime.h>
#include <hip/hip_bf16.h>
#include <math.h>

// Problem constants (from reference): T=512, B=128, H=512
#define TT 512
#define BB 128
#define HH 512
#define H3 (3 * HH)

// LDS weight cache, chunk-major: 16 K-chunks * 6 (matrix,gate) planes * 64
// lane-fragments * 32 bytes = 192 KB (CDNA5 allows up to 320 KB per WG).
// Within a chunk the 6 planes span 12 KB -> all DS loads use 16-bit
// immediate offsets from one per-chunk base.
#define NFRAG   (16 * 6 * 64)
#define LDSB    (NFRAG * 32)

typedef __attribute__((ext_vector_type(16))) _Float16 v16h;
typedef __attribute__((ext_vector_type(8)))  _Float16 v8h;
typedef __attribute__((ext_vector_type(8)))  float    v8f;

// ---------------------------------------------------------------------------
// Device-scope ticket barrier (64 blocks, trivially grid-resident on MI455X).
// Counter zeroed by the init kernel each launch; 512 uses * 64 blocks max.
// ---------------------------------------------------------------------------
__device__ __forceinline__ void grid_barrier(unsigned* cnt, unsigned nblk) {
    __syncthreads();
    __threadfence();
    if (threadIdx.x == 0) {
        unsigned ticket = __hip_atomic_fetch_add(cnt, 1u, __ATOMIC_ACQ_REL,
                                                 __HIP_MEMORY_SCOPE_AGENT);
        unsigned target = (ticket / nblk + 1u) * nblk;
        while (__hip_atomic_load(cnt, __ATOMIC_ACQUIRE,
                                 __HIP_MEMORY_SCOPE_AGENT) < target) {
            __builtin_amdgcn_s_sleep(8);
        }
    }
    __syncthreads();
}

__device__ __forceinline__ float sigmoidf_(float v) {
    return 1.0f / (1.0f + __expf(-v));
}

__device__ __forceinline__ v8f wmma_f16(v16h a, v16h b, v8f c) {
    return __builtin_amdgcn_wmma_f32_16x16x32_f16(
        /*neg_a=*/false, a, /*neg_b=*/false, b,
        /*c_mod=*/(short)0, c, /*reuse_a=*/false, /*reuse_b=*/false);
}

// ---------------------------------------------------------------------------
// Init: zero barrier counter; build masked step-0 carry (f32 + f16); convert
// x[0] into xbuf parity-0. resets is a jax bool array -> 1 byte per element.
// ---------------------------------------------------------------------------
__global__ void gru_init_kernel(const float* __restrict__ x,
                                const unsigned char* __restrict__ resets,
                                const float* __restrict__ h0,
                                _Float16* __restrict__ xbuf,
                                _Float16* __restrict__ hf16,
                                float* __restrict__ hf32,
                                unsigned* __restrict__ cnt) {
    int gtid = blockIdx.x * blockDim.x + threadIdx.x;   // 0 .. B*H-1
    if (gtid == 0) *cnt = 0u;
    if (gtid < BB * HH) {
        int m = gtid / HH;
        float hm = resets[m] ? 0.0f : h0[gtid];
        hf32[gtid] = hm;
        hf16[gtid] = (_Float16)hm;
        xbuf[gtid] = (_Float16)x[gtid];      // x[t=0] -> parity-0 buffer
    }
}

// ---------------------------------------------------------------------------
// Weight transpose+convert: W[H][3H] fp32 -> Wt[3H][H] f16 (lane-contiguous
// K for WMMA B-fragments).
// ---------------------------------------------------------------------------
__global__ void gru_wconv_kernel(const float* __restrict__ Wi,
                                 const float* __restrict__ Wh,
                                 _Float16* __restrict__ WtI,
                                 _Float16* __restrict__ WtH) {
    long e = (long)blockIdx.x * blockDim.x + threadIdx.x;  // 0 .. 2*H3*H-1
    const long half = (long)H3 * HH;
    if (e >= 2 * half) return;
    const float* src = (e < half) ? Wi : Wh;
    _Float16*    dst = (e < half) ? WtI : WtH;
    long r = (e < half) ? e : e - half;
    int c = (int)(r / HH);   // 0..3H-1 : output row = column of W
    int k = (int)(r % HH);   // 0..H-1  : output col = row of W
    dst[r] = (_Float16)src[(long)k * H3 + c];
}

// ---------------------------------------------------------------------------
// Persistent scan. Grid (16,4) blocks of 128 threads (4 waves = 2x2 tiles of
// 16x16): blockIdx.x -> 32 H-columns, blockIdx.y -> 32 batch rows.
// Weights for the block's 32 columns (3 gates, both matrices, full K) are
// staged once into LDS as pre-swizzled 32B lane-fragments, chunk-major.
// The K-loop is a manually double-buffered software pipeline, fully unrolled:
// chunk kc+1's 12 ds_load_b128 + 4 global_load_b128 issue before chunk kc's
// 6 WMMAs, hiding LDS/L2 latency behind matrix issue. launch_bounds(...,1)
// (1 wave/SIMD — forced anyway by 192 KB LDS) unlocks the full VGPR file so
// the two fragment register sets don't get collapsed.
// One grid barrier per step; x[t+1] converted during step t (double-buffered).
// ---------------------------------------------------------------------------
__global__ void __launch_bounds__(128, 1)
gru_scan_kernel(const float* __restrict__ x,
                const unsigned char* __restrict__ resets,
                const float* __restrict__ bi,
                const float* __restrict__ bhn,
                const _Float16* __restrict__ WtI,
                const _Float16* __restrict__ WtH,
                _Float16* __restrict__ xbuf,      // [2][B][H] f16
                _Float16* __restrict__ hf16,      // [2][B][H] f16
                float* __restrict__ hf32,         // [2][B][H] f32
                float* __restrict__ ys,           // [T][B][H] f32 (d_out)
                unsigned* __restrict__ cnt) {
    extern __shared__ _Float16 smem[];            // 192 KB weight cache

    const unsigned nblk = gridDim.x * gridDim.y;
    const int tid  = threadIdx.x;
    const int bx   = blockIdx.x;                  // 0..15 column tile
    const int by   = blockIdx.y;                  // 0..3  row tile
    const int bid  = by * gridDim.x + bx;
    const int gtid = bid * (int)blockDim.x + tid; // 0 .. 8191

    const int lane = tid & 31;
    const int wave = tid >> 5;
    const int wn   = wave & 1;                    // column sub-tile
    const int wm   = wave >> 1;                   // row sub-tile
    const int hi   = lane >> 4;                   // K-half selector
    const int mr   = lane & 15;
    const int m0   = by * 32 + wm * 16;
    const int col  = bx * 32 + wn * 16 + mr;      // lane's output column

    // ---- Stage weight slice into LDS (once, reused for all 512 steps).
    // Chunk-major fragment index: (kc*6 + wg)*64 + (wn_f*32 + lane_f).
    for (int idx = tid; idx < NFRAG; idx += 128) {
        int fi   = idx & 63;
        int rest = idx >> 6;                      // 0..95
        int wg   = rest % 6;                      // w*3+g
        int kc   = rest / 6;
        int w    = wg / 3;
        int g    = wg % 3;
        int lf   = fi & 31;
        int colf = bx * 32 + (fi >> 5) * 16 + (lf & 15);
        const _Float16* src = (w == 0 ? WtI : WtH)
                            + (long)(g * HH + colf) * HH + kc * 32 + (lf >> 4) * 16;
        *(v16h*)(smem + (long)idx * 16) = *(const v16h*)src;
    }
    __syncthreads();

    // Per-lane LDS fragment base. Chunk stride = 6*64*16 halves (12 KB);
    // plane stride within a chunk = 64*16 = 1024 halves (2 KB immediates).
    const _Float16* lwfrag = smem + (wn * 32 + lane) * 16;
#define KCHUNK 6144
#define PLANE(w, g) (((w) * 3 + (g)) * 1024)

    const long arow = (long)(m0 + mr) * HH + hi * 16;   // A-fragment offset

    const float b_r  = bi[col];
    const float b_z  = bi[HH + col];
    const float b_n  = bi[2 * HH + col];
    const float b_hn = bhn[col];

    for (int t = 0; t < TT; ++t) {
        // ---- Convert x[t+1] -> xbuf[(t+1)&1] (overlapped with compute) ----
        if (t + 1 < TT) {
            const float4* xt = (const float4*)(x + (long)(t + 1) * (BB * HH));
            float4 a = xt[2 * gtid];
            float4 b = xt[2 * gtid + 1];
            v8h o;
            o[0] = (_Float16)a.x; o[1] = (_Float16)a.y;
            o[2] = (_Float16)a.z; o[3] = (_Float16)a.w;
            o[4] = (_Float16)b.x; o[5] = (_Float16)b.y;
            o[6] = (_Float16)b.z; o[7] = (_Float16)b.w;
            ((v8h*)(xbuf + (long)((t + 1) & 1) * (BB * HH)))[gtid] = o;
        }

        // ---- Fused (x@Wi + h@Wh) WMMA GEMMs, software-pipelined ----
        const _Float16* xrow = xbuf + (long)(t & 1) * (BB * HH) + arow;
        const _Float16* hrow = hf16 + (long)(t & 1) * (BB * HH) + arow;
        const float*    hc32 = hf32 + (long)(t & 1) * (BB * HH);
        _Float16*       hn16 = hf16 + (long)((t + 1) & 1) * (BB * HH);
        float*          hn32 = hf32 + (long)((t + 1) & 1) * (BB * HH);

        v8f rc = {}, zc = {}, nc = {}, hc = {};

        // Prologue: load chunk 0 into the compute set.
        v16h ax_c = *(const v16h*)(xrow);
        v16h ah_c = *(const v16h*)(hrow);
        v16h b0_c = *(const v16h*)(lwfrag + PLANE(0, 0));
        v16h b1_c = *(const v16h*)(lwfrag + PLANE(0, 1));
        v16h b2_c = *(const v16h*)(lwfrag + PLANE(0, 2));
        v16h b3_c = *(const v16h*)(lwfrag + PLANE(1, 0));
        v16h b4_c = *(const v16h*)(lwfrag + PLANE(1, 1));
        v16h b5_c = *(const v16h*)(lwfrag + PLANE(1, 2));

        #pragma unroll
        for (int kc = 0; kc < 16; ++kc) {
            v16h ax_n, ah_n, b0_n, b1_n, b2_n, b3_n, b4_n, b5_n;
            if (kc < 15) {
                // Issue next chunk's loads before this chunk's WMMAs.
                const _Float16* ln = lwfrag + (kc + 1) * KCHUNK;
                ax_n = *(const v16h*)(xrow + (kc + 1) * 32);
                ah_n = *(const v16h*)(hrow + (kc + 1) * 32);
                b0_n = *(const v16h*)(ln + PLANE(0, 0));
                b1_n = *(const v16h*)(ln + PLANE(0, 1));
                b2_n = *(const v16h*)(ln + PLANE(0, 2));
                b3_n = *(const v16h*)(ln + PLANE(1, 0));
                b4_n = *(const v16h*)(ln + PLANE(1, 1));
                b5_n = *(const v16h*)(ln + PLANE(1, 2));
            }
            // Rotate accumulators so consecutive WMMAs are independent.
            rc = wmma_f16(ax_c, b0_c, rc);
            zc = wmma_f16(ax_c, b1_c, zc);
            nc = wmma_f16(ax_c, b2_c, nc);
            rc = wmma_f16(ah_c, b3_c, rc);
            zc = wmma_f16(ah_c, b4_c, zc);
            hc = wmma_f16(ah_c, b5_c, hc);
            if (kc < 15) {
                ax_c = ax_n; ah_c = ah_n;
                b0_c = b0_n; b1_c = b1_n; b2_c = b2_n;
                b3_c = b3_n; b4_c = b4_n; b5_c = b5_n;
            }
        }

        // ---- Gating. C/D layout: VGPR j -> M = j + hi*8, N = mr. ----
        #pragma unroll
        for (int j = 0; j < 8; ++j) {
            int m = m0 + hi * 8 + j;
            float r = sigmoidf_(rc[j] + b_r);
            float z = sigmoidf_(zc[j] + b_z);
            float n = tanhf(nc[j] + b_n + r * (hc[j] + b_hn));
            float hp = hc32[(long)m * HH + col];
            float hnew = (1.0f - z) * n + z * hp;
            ys[((long)t * BB + m) * HH + col] = hnew;
            float hm = hnew;
            if (t + 1 < TT) hm = resets[(long)(t + 1) * BB + m] ? 0.0f : hnew;
            hn32[(long)m * HH + col] = hm;
            hn16[(long)m * HH + col] = (_Float16)hm;
        }

        grid_barrier(cnt, nblk);   // single barrier per step
    }
#undef PLANE
#undef KCHUNK
}

// ---------------------------------------------------------------------------
// Host launcher. Inputs: x, resets, Wi, bi, Wh, bhn, h0 (setup_inputs order).
// Scratch (~4 MB), all regions 256B aligned.
// ---------------------------------------------------------------------------
extern "C" void kernel_launch(void* const* d_in, const int* in_sizes, int n_in,
                              void* d_out, int out_size, void* d_ws, size_t ws_size,
                              hipStream_t stream) {
    const float*         x      = (const float*)d_in[0];
    const unsigned char* resets = (const unsigned char*)d_in[1];  // jax bool = 1B
    const float*         Wi     = (const float*)d_in[2];
    const float*         bi     = (const float*)d_in[3];
    const float*         Wh     = (const float*)d_in[4];
    const float*         bhn    = (const float*)d_in[5];
    const float*         h0     = (const float*)d_in[6];
    float*               ys     = (float*)d_out;

    char* p = (char*)d_ws;
    unsigned* cnt  = (unsigned*)p;      p += 256;
    _Float16* xbuf = (_Float16*)p;      p += 2L * BB * HH * 2;         // 256 KB
    _Float16* hf16 = (_Float16*)p;      p += 2L * BB * HH * 2;         // 256 KB
    float*    hf32 = (float*)p;         p += 2L * BB * HH * 4;         // 512 KB
    _Float16* WtI  = (_Float16*)p;      p += (long)H3 * HH * 2;        // 1.5 MB
    _Float16* WtH  = (_Float16*)p;      p += (long)H3 * HH * 2;        // 1.5 MB

    // 1) barrier counter + masked initial carry + x[0] conversion
    gru_init_kernel<<<dim3((BB * HH + 255) / 256), dim3(256), 0, stream>>>(
        x, resets, h0, xbuf, hf16, hf32, cnt);

    // 2) weight transpose+convert to f16 [3H][H]
    {
        long total = 2L * H3 * HH;
        gru_wconv_kernel<<<dim3((unsigned)((total + 255) / 256)), dim3(256), 0,
                           stream>>>(Wi, Wh, WtI, WtH);
    }

    // 3) persistent scan: 64 blocks (grid-resident), 128 threads, 192KB LDS
    gru_scan_kernel<<<dim3(16, 4), dim3(128), LDSB, stream>>>(
        x, resets, bi, bhn, WtI, WtH, xbuf, hf16, hf32, ys, cnt);
}
